// GATconv_90615220011131
// MI455X (gfx1250) — compile-verified, matrix operand order checked by reading, not run on hardware
//
#include <hip/hip_runtime.h>

typedef __attribute__((ext_vector_type(2)))  float    v2f;
typedef __attribute__((ext_vector_type(8)))  float    v8f;
typedef __attribute__((ext_vector_type(16))) _Float16 v16h;
typedef __attribute__((ext_vector_type(4)))  unsigned v4u;
typedef __attribute__((ext_vector_type(8)))  int      v8i;
typedef __attribute__((ext_vector_type(4)))  int      v4i;

// ---- monotone float<->uint key for atomicMax-based segment max ----
__device__ __forceinline__ unsigned fkey(float f) {
  unsigned b = __float_as_uint(f);
  return (b & 0x80000000u) ? ~b : (b | 0x80000000u);
}
__device__ __forceinline__ float funkey(unsigned k) {
  unsigned b = (k & 0x80000000u) ? (k & 0x7fffffffu) : ~k;
  return __uint_as_float(b);
}
__device__ __forceinline__ float lrelu(float a) { return a > 0.f ? a : 0.2f * a; }

// ---------------- K0: zero out / mkey / ssum ----------------
__global__ void k_init(float* __restrict__ out, unsigned* __restrict__ mkey,
                       float* __restrict__ ssum, int n) {
  int i = blockIdx.x * blockDim.x + threadIdx.x;
  if (i < n * 128) out[i] = 0.f;
  if (i < n * 4) { mkey[i] = 0u; ssum[i] = 0.f; }   // key 0 acts as -inf (self loops guarantee hits)
}

// ---------------- K1: h = x @ W^T  (fp32 WMMA 16x16x4, A tile staged via TDM) ----------------
__global__ void __launch_bounds__(256) k_gemm(const float* __restrict__ x,
                                              const float* __restrict__ W,
                                              float* __restrict__ h, int n) {
  __shared__ float As[16 * 132];                 // row stride 132 dwords (bank spread)
  const int m0   = blockIdx.x * 16;
  const int wave = threadIdx.x >> 5;             // 8 waves -> 8 column tiles of 16
  const int lane = threadIdx.x & 31;
  const int l    = lane & 15;
  const int half = lane >> 4;
  const int n0   = wave * 16;

#if __has_builtin(__builtin_amdgcn_tensor_load_to_lds)
  // Tensor Data Mover: one DMA of the 16x128 fp32 A tile, LDS-padded 4 dwords
  // per 128 stored (-> stride 132), issued once by wave 0.
  if (threadIdx.x == 0) {
    const unsigned rows   = (unsigned)((n - m0) < 16 ? (n - m0) : 16);
    const unsigned ldsoff = (unsigned)(size_t)(&As[0]);   // flat LDS addr low 32 = offset
    const unsigned long long ga = (unsigned long long)(const void*)(x + (size_t)m0 * 128);
    v4u g0;
    g0[0] = 1u;                                            // count=1 (valid user D#)
    g0[1] = ldsoff;                                        // lds_addr
    g0[2] = (unsigned)ga;                                  // global_addr[31:0]
    g0[3] = (unsigned)((ga >> 32) & 0x01ffffffu) | (2u << 30); // global_addr[56:32], type=2
    v8i g1;
    g1[0] = (int)((2u << 16)     // data_size = 4B
                | (1u << 20)     // pad_enable
                | (6u << 22)     // pad_interval: every 128 dwords
                | (3u << 25));   // pad_amount: 4 dwords
    g1[1] = (int)(128u << 16);                 // tensor_dim0 = 128 (bits 79:48 lo)
    g1[2] = (int)(rows << 16);                 // tensor_dim0 hi=0 | tensor_dim1 = rows
    g1[3] = (int)(128u << 16);                 // tensor_dim1 hi=0 | tile_dim0 = 128
    g1[4] = 16;                                // tile_dim1 = 16 | tile_dim2 = 0
    g1[5] = 128;                               // tensor_dim0_stride = 128 (lo32)
    g1[6] = 0;                                 // stride hi | tensor_dim1_stride lo
    g1[7] = 0;
    v4i gz4 = {0, 0, 0, 0};
    v8i gz8 = {0, 0, 0, 0, 0, 0, 0, 0};
    __builtin_amdgcn_tensor_load_to_lds(g0, g1, gz4, gz4, gz8, 0);   // 6-arg toolchain form
    __builtin_amdgcn_s_wait_tensorcnt(0);      // s_wait_tensorcnt 0
  }
  __syncthreads();
#else
  for (int t = threadIdx.x; t < 16 * 128; t += 256) {
    int r = t >> 7, c = t & 127;
    int row = m0 + r;
    As[r * 132 + c] = (row < n) ? x[(size_t)row * 128 + c] : 0.f;
  }
  __syncthreads();
#endif

  const float* __restrict__ arow = &As[l * 132];               // A row m0+l
  const float* __restrict__ brow = W + (size_t)(n0 + l) * 128; // B col n0+l == W row
  v8f acc = {};

#if __has_builtin(__builtin_amdgcn_wmma_f32_16x16x4_f32)
  // A 16x4 f32: lane l<16 holds (M=l, K=2*half+{0,1}) per ISA layout
  for (int kb = 0; kb < 32; ++kb) {
    const int k = kb * 4 + half * 2;
    v2f a, b;
    a.x = arow[k]; a.y = arow[k + 1];
    b.x = brow[k]; b.y = brow[k + 1];
    acc = __builtin_amdgcn_wmma_f32_16x16x4_f32(false, a, false, b,
                                                (short)0, acc, false, false);
  }
#else
  for (int kc = 0; kc < 128; kc += 32) {
    v16h a, b;
    for (int j = 0; j < 16; ++j) {
      int k = (j < 8) ? (half * 8 + j) : (16 + half * 8 + (j - 8));
      a[j] = (_Float16)arow[kc + k];
      b[j] = (_Float16)brow[kc + k];
    }
    acc = __builtin_amdgcn_wmma_f32_16x16x32_f16(false, a, false, b,
                                                 (short)0, acc, false, false);
  }
#endif

  // C/D layout: VGPR i -> M = i (lanes 0-15) / 8+i (lanes 16-31), N = lane%16
  for (int i = 0; i < 8; ++i) {
    int row = m0 + (half ? 8 + i : i);
    if (row < n) h[(size_t)row * 128 + n0 + l] = acc[i];
  }
}

// ---------------- K2: per-node per-head attention logits ----------------
__global__ void k_logits(const float* __restrict__ h,
                         const float* __restrict__ att_src,
                         const float* __restrict__ att_dst,
                         float* __restrict__ a_src, float* __restrict__ a_dst, int n) {
  int i = blockIdx.x * blockDim.x + threadIdx.x;
  if (i >= n * 4) return;
  int node = i >> 2, hd = i & 3;
  const float* hp = h + (size_t)node * 128 + hd * 32;
  const float* as = att_src + hd * 32;
  const float* ad = att_dst + hd * 32;
  float s1 = 0.f, s2 = 0.f;
#pragma unroll
  for (int d = 0; d < 32; ++d) {
    float v = hp[d];
    s1 += v * as[d];
    s2 += v * ad[d];
  }
  a_src[i] = s1;
  a_dst[i] = s2;
}

// ---------------- K3: segment max over dst (ordered-uint atomicMax) ----------------
__global__ void k_edge_max(const int* __restrict__ ei,
                           const float* __restrict__ a_src,
                           const float* __restrict__ a_dst,
                           unsigned* __restrict__ mkey, int E, int n) {
  int e = blockIdx.x * blockDim.x + threadIdx.x;
  if (e >= E + n) return;
  int src, dst;
  if (e < E) { src = ei[e]; dst = ei[E + e]; } else { src = dst = e - E; }
  const float4 as4 = *(const float4*)(a_src + (size_t)src * 4);  // one b128 gather
  const float4 ad4 = *(const float4*)(a_dst + (size_t)dst * 4);
  const float als[4] = {as4.x + ad4.x, as4.y + ad4.y, as4.z + ad4.z, as4.w + ad4.w};
  unsigned* mk = mkey + (size_t)dst * 4;
#pragma unroll
  for (int hd = 0; hd < 4; ++hd) atomicMax(&mk[hd], fkey(lrelu(als[hd])));
}

// ---------------- K4: segment sum of exp(alpha - m) ----------------
__global__ void k_edge_sum(const int* __restrict__ ei,
                           const float* __restrict__ a_src,
                           const float* __restrict__ a_dst,
                           const unsigned* __restrict__ mkey,
                           float* __restrict__ ssum, int E, int n) {
  int e = blockIdx.x * blockDim.x + threadIdx.x;
  if (e >= E + n) return;
  int src, dst;
  if (e < E) { src = ei[e]; dst = ei[E + e]; } else { src = dst = e - E; }
  const float4 as4 = *(const float4*)(a_src + (size_t)src * 4);
  const float4 ad4 = *(const float4*)(a_dst + (size_t)dst * 4);
  const uint4  mk4 = *(const uint4*)(mkey + (size_t)dst * 4);
  const float als[4] = {as4.x + ad4.x, as4.y + ad4.y, as4.z + ad4.z, as4.w + ad4.w};
  const unsigned mks[4] = {mk4.x, mk4.y, mk4.z, mk4.w};
  float* sp = ssum + (size_t)dst * 4;
#pragma unroll
  for (int hd = 0; hd < 4; ++hd)
    atomicAdd(&sp[hd], __expf(lrelu(als[hd]) - funkey(mks[hd])));
}

// ---------------- K5: weighted scatter-add, one wave32 per edge ----------------
__global__ void __launch_bounds__(256) k_aggregate(const int* __restrict__ ei,
                                                   const float* __restrict__ h,
                                                   const float* __restrict__ a_src,
                                                   const float* __restrict__ a_dst,
                                                   const unsigned* __restrict__ mkey,
                                                   const float* __restrict__ ssum,
                                                   float* __restrict__ out, int E, int n) {
  int e    = (int)((blockIdx.x * blockDim.x + threadIdx.x) >> 5);
  int lane = threadIdx.x & 31;
  if (e >= E + n) return;
  int src, dst;
  if (e < E) { src = ei[e]; dst = ei[E + e]; } else { src = dst = e - E; }

  int hd   = lane >> 3;                                // lane's head (4 floats/lane, 32 dims/head)
  float al = lrelu(a_src[src * 4 + hd] + a_dst[dst * 4 + hd]);
  float m  = funkey(mkey[dst * 4 + hd]);
  float w  = __expf(al - m) / ssum[dst * 4 + hd] + 1.0f;   // f_additive: e + 1

  const float4 hv = *(const float4*)(h + (size_t)src * 128 + lane * 4);
  float* op = out + (size_t)dst * 128 + lane * 4;
  atomicAdd(op + 0, hv.x * w);
  atomicAdd(op + 1, hv.y * w);
  atomicAdd(op + 2, hv.z * w);
  atomicAdd(op + 3, hv.w * w);
}

extern "C" void kernel_launch(void* const* d_in, const int* in_sizes, int n_in,
                              void* d_out, int out_size, void* d_ws, size_t ws_size,
                              hipStream_t stream) {
  const float* x       = (const float*)d_in[0];
  const int*   ei      = (const int*)d_in[1];
  const float* W       = (const float*)d_in[2];
  const float* att_src = (const float*)d_in[3];
  const float* att_dst = (const float*)d_in[4];
  float* out = (float*)d_out;

  const int n  = in_sizes[0] / 128;
  const int E  = in_sizes[1] / 2;
  const int ET = E + n;

  float*    h     = (float*)d_ws;                      // n*128
  float*    a_src = h + (size_t)n * 128;               // n*4
  float*    a_dst = a_src + (size_t)n * 4;             // n*4
  unsigned* mkey  = (unsigned*)(a_dst + (size_t)n * 4);// n*4
  float*    ssum  = (float*)(mkey + (size_t)n * 4);    // n*4

  k_init     <<<(n * 128 + 255) / 256, 256, 0, stream>>>(out, mkey, ssum, n);
  k_gemm     <<<(n + 15) / 16,          256, 0, stream>>>(x, W, h, n);
  k_logits   <<<(n * 4 + 255) / 256,    256, 0, stream>>>(h, att_src, att_dst, a_src, a_dst, n);
  k_edge_max <<<(ET + 255) / 256,       256, 0, stream>>>(ei, a_src, a_dst, mkey, E, n);
  k_edge_sum <<<(ET + 255) / 256,       256, 0, stream>>>(ei, a_src, a_dst, mkey, ssum, E, n);
  k_aggregate<<<(ET + 7) / 8,           256, 0, stream>>>(ei, h, a_src, a_dst, mkey, ssum, out, E, n);
}
